// EncoderBlock_35081292874276
// MI455X (gfx1250) — matrix operand order repaired and checked by reading, not compiled, not run on hardware
//
#include <hip/hip_runtime.h>

// ---------------- CDNA5 (gfx1250) types ----------------
typedef __bf16 bf16_t;
typedef __bf16 v16bf __attribute__((ext_vector_type(16)));
typedef __bf16 v8bf  __attribute__((ext_vector_type(8)));
typedef float  v8f   __attribute__((ext_vector_type(8)));

union Frag16 { v16bf v; v8bf h[2]; };

#define WMMA_BF16(A_, B_, C_) \
  __builtin_amdgcn_wmma_f32_16x16x32_bf16(false, (A_), false, (B_), (short)0, (C_), false, false)

// ---------------- problem constants ----------------
#define NB   2
#define SEQ  2048
#define DM   1024
#define NH   16
#define DH   64
#define FF   4096

// ============================================================
// f32 -> bf16 conversion (grid-stride)
// ============================================================
__global__ void eb_cvt_bf16(const float* __restrict__ in, bf16_t* __restrict__ out, int n) {
  int i = blockIdx.x * blockDim.x + threadIdx.x;
  int stride = gridDim.x * blockDim.x;
  for (; i < n; i += stride) out[i] = (bf16_t)in[i];
}

// ============================================================
// Tiled bf16 GEMM with WMMA. Block tile 128xBN (BN = 64 or 128), BK = 32.
// 256 threads = 8 waves, wave grid 4x2; each wave 32 x (BN/2):
//   BN=64  -> 2x2 WMMA tiles,  4 WMMA / K-step / wave
//   BN=128 -> 2x4 WMMA tiles,  8 WMMA / K-step / wave (2x A reuse)
// C = A[128xK] * B[KxBN] + bias; epilogue: optional relu, f32 or bf16 out.
// Batched over gridDim.z: A index = z/zDivA, B/bias index = z%zModB, C index = z.
// All global staging pointers are strength-reduced (no 64-bit mul in loop).
// ============================================================
template<int BN>
__global__ __launch_bounds__(256)
void eb_gemm(const bf16_t* __restrict__ A, long sAz, int zDivA, int lda,
             const bf16_t* __restrict__ B, long sBz, int zModB, int ldb,
             const float*  __restrict__ bias, long sBiasz,
             float* __restrict__ Cf, bf16_t* __restrict__ Cb, long sCz, int ldc,
             int Kdim, int relu)
{
  constexpr int NT  = BN / 32;   // WMMA col tiles per wave
  constexpr int NBB = BN / 64;   // B staging chunks per thread

  const int z = blockIdx.z;
  A    += (long)(z / zDivA) * sAz;
  B    += (long)(z % zModB) * sBz;
  bias += (long)(z % zModB) * sBiasz;
  const long cOff = (long)z * sCz;

  __shared__ __align__(16) bf16_t As[128][32];   // 8 KB
  __shared__ __align__(16) bf16_t Bt[BN][32];    // 4/8 KB (transposed: Bt[n][k])

  const int tid  = threadIdx.x;
  const int lane = tid & 31;
  const int wid  = tid >> 5;
  const int wm   = wid & 3;    // 4 row-waves (32 rows each)
  const int wn   = wid >> 2;   // 2 col-waves (BN/2 cols each)
  const int r    = lane & 15;
  const int hi   = lane >> 4;

  const long rowBase = (long)blockIdx.x * 128;
  const int  colBase = blockIdx.y * BN;

  // ---- strength-reduced global staging pointers ----
  const int ar0 = tid >> 2;              // A: chunk tid -> row, and row+64 for 2nd chunk
  const int ac0 = (tid & 3) << 3;
  const bf16_t* aG0 = A + (rowBase + ar0)      * (long)lda + ac0;
  const bf16_t* aG1 = A + (rowBase + ar0 + 64) * (long)lda + ac0;

  const int kk  = tid >> 3;              // B: 0..31 (K row within tile)
  const int bn0 = (tid & 7) << 3;        //    col chunk of 8
  const bf16_t* bG[NBB];
#pragma unroll
  for (int bb = 0; bb < NBB; ++bb)
    bG[bb] = B + (long)kk * ldb + colBase + bn0 + bb * 64;
  const long bStep = 32 * (long)ldb;

  v8f acc[2][NT];
#pragma unroll
  for (int mi = 0; mi < 2; ++mi)
#pragma unroll
    for (int ni = 0; ni < NT; ++ni)
#pragma unroll
      for (int e = 0; e < 8; ++e) acc[mi][ni][e] = 0.f;

  for (int k0 = 0; k0 < Kdim; k0 += 32) {
    __syncthreads();
    // A tile 128x32: two b128 global->LDS chunks per thread
    *(v8bf*)&As[ar0][ac0]      = *(const v8bf*)aG0;  aG0 += 32;
    *(v8bf*)&As[ar0 + 64][ac0] = *(const v8bf*)aG1;  aG1 += 32;
    // B tile 32xBN, written transposed
#pragma unroll
    for (int bb = 0; bb < NBB; ++bb) {
      v8bf tb = *(const v8bf*)bG[bb];  bG[bb] += bStep;
#pragma unroll
      for (int j = 0; j < 8; ++j) Bt[bn0 + bb * 64 + j][kk] = tb[j];
    }
    __syncthreads();

    Frag16 af[2], bfr[NT];
#pragma unroll
    for (int mi = 0; mi < 2; ++mi) {
      int rl = wm * 32 + mi * 16 + r;
      af[mi].h[0] = *(const v8bf*)&As[rl][hi * 8];
      af[mi].h[1] = *(const v8bf*)&As[rl][16 + hi * 8];
    }
#pragma unroll
    for (int ni = 0; ni < NT; ++ni) {
      int cl = wn * (BN / 2) + ni * 16 + r;
      bfr[ni].h[0] = *(const v8bf*)&Bt[cl][hi * 8];
      bfr[ni].h[1] = *(const v8bf*)&Bt[cl][16 + hi * 8];
    }
#pragma unroll
    for (int mi = 0; mi < 2; ++mi)
#pragma unroll
      for (int ni = 0; ni < NT; ++ni)
        acc[mi][ni] = WMMA_BF16(af[mi].v, bfr[ni].v, acc[mi][ni]);
  }

  // epilogue: C layout — lane holds col (ni*16 + r), rows (mi*16 + hi*8 + e)
#pragma unroll
  for (int mi = 0; mi < 2; ++mi) {
#pragma unroll
    for (int ni = 0; ni < NT; ++ni) {
      int col = colBase + wn * (BN / 2) + ni * 16 + r;
      float bv = bias[col];
      long rowG = rowBase + wm * 32 + mi * 16 + hi * 8;
#pragma unroll
      for (int e = 0; e < 8; ++e) {
        float xv = acc[mi][ni][e] + bv;
        if (relu) xv = fmaxf(xv, 0.f);
        long idx = cOff + (rowG + e) * (long)ldc + col;
        if (Cb) Cb[idx] = (bf16_t)xv; else Cf[idx] = xv;
      }
    }
  }
}

// ============================================================
// Flash-attention style streaming softmax attention.
// grid: (SEQ/128, NB*NH). 256 threads = 8 waves; wave owns 16 queries.
// q/k/v: [NB*NH][SEQ][DH] bf16. Output written into cat [NB*SEQ][DM] bf16
// at head column offset h*DH (concat-heads layout for the Wo GEMM).
// ============================================================
__global__ __launch_bounds__(256)
void eb_flash(const bf16_t* __restrict__ q, const bf16_t* __restrict__ kmat,
              const bf16_t* __restrict__ v, bf16_t* __restrict__ cat)
{
  const int z = blockIdx.y;           // n*NH + h
  const int n = z / NH, h = z % NH;
  const long base = (long)z * SEQ * DH;
  const bf16_t* qb = q + base;

  const int tid = threadIdx.x, lane = tid & 31, wid = tid >> 5;
  const int r = lane & 15, hi = lane >> 4;
  const int q0 = blockIdx.x * 128 + wid * 16;

  __shared__ __align__(16) bf16_t Ks[32][64];     // key tile, row-major (== Bt for Q*K^T)
  __shared__ __align__(16) bf16_t Vt[64][32];     // V tile transposed (Bt for P*V)
  __shared__ __align__(16) bf16_t Ps[8][16][32];  // per-wave P relayout buffer

  // Q A-fragments for d-chunks {0..31, 32..63} — loop invariant
  Frag16 qf[2];
#pragma unroll
  for (int c = 0; c < 2; ++c) {
    const bf16_t* p = qb + (long)(q0 + r) * DH + c * 32;
    qf[c].h[0] = *(const v8bf*)(p + hi * 8);
    qf[c].h[1] = *(const v8bf*)(p + 16 + hi * 8);
  }

  // strength-reduced K/V staging pointers
  const int kr = tid >> 3, c0 = (tid & 7) << 3;
  const bf16_t* kG = kmat + base + (long)kr * DH + c0;
  const bf16_t* vG = v    + base + (long)kr * DH + c0;

  float m[8], l[8];
  v8f o[4];
#pragma unroll
  for (int i = 0; i < 8; ++i) { m[i] = -1e30f; l[i] = 0.f; }
#pragma unroll
  for (int ct = 0; ct < 4; ++ct)
#pragma unroll
    for (int i = 0; i < 8; ++i) o[ct][i] = 0.f;

  for (int key0 = 0; key0 < SEQ; key0 += 32) {
    __syncthreads();
    {   // stage K row-major, V transposed
      *(v8bf*)&Ks[kr][c0] = *(const v8bf*)kG;  kG += 32 * DH;
      v8bf tv = *(const v8bf*)vG;              vG += 32 * DH;
#pragma unroll
      for (int j = 0; j < 8; ++j) Vt[c0 + j][kr] = tv[j];
    }
    __syncthreads();

    // S[16 queries x 32 keys] = Q * K^T, scale 1/sqrt(64)
    v8f s[2];
#pragma unroll
    for (int t = 0; t < 2; ++t) {
#pragma unroll
      for (int i = 0; i < 8; ++i) s[t][i] = 0.f;
#pragma unroll
      for (int c = 0; c < 2; ++c) {
        Frag16 kf;
        kf.h[0] = *(const v8bf*)&Ks[t * 16 + r][c * 32 + hi * 8];
        kf.h[1] = *(const v8bf*)&Ks[t * 16 + r][c * 32 + 16 + hi * 8];
        s[t] = WMMA_BF16(qf[c].v, kf.v, s[t]);
      }
#pragma unroll
      for (int i = 0; i < 8; ++i) s[t][i] *= 0.125f;
    }

    // online softmax: row-wise max/sum across the 16-lane column groups
    float rmax[8], rsum[8], p0[8], p1[8];
#pragma unroll
    for (int i = 0; i < 8; ++i) rmax[i] = fmaxf(s[0][i], s[1][i]);
#pragma unroll
    for (int off = 1; off < 16; off <<= 1)
#pragma unroll
      for (int i = 0; i < 8; ++i) rmax[i] = fmaxf(rmax[i], __shfl_xor(rmax[i], off, 32));
#pragma unroll
    for (int i = 0; i < 8; ++i) {
      float mn = fmaxf(m[i], rmax[i]);
      float al = __expf(m[i] - mn);
      p0[i] = __expf(s[0][i] - mn);
      p1[i] = __expf(s[1][i] - mn);
      m[i] = mn;
      l[i] *= al;
#pragma unroll
      for (int ct = 0; ct < 4; ++ct) o[ct][i] *= al;
      rsum[i] = p0[i] + p1[i];
    }
#pragma unroll
    for (int off = 1; off < 16; off <<= 1)
#pragma unroll
      for (int i = 0; i < 8; ++i) rsum[i] += __shfl_xor(rsum[i], off, 32);
#pragma unroll
    for (int i = 0; i < 8; ++i) l[i] += rsum[i];

    // re-layout P (C-layout) -> A-fragment via per-wave LDS (DS ops are in-order)
#pragma unroll
    for (int i = 0; i < 8; ++i) {
      Ps[wid][hi * 8 + i][r]      = (bf16_t)p0[i];
      Ps[wid][hi * 8 + i][16 + r] = (bf16_t)p1[i];
    }
    Frag16 pf;
    pf.h[0] = *(const v8bf*)&Ps[wid][r][hi * 8];
    pf.h[1] = *(const v8bf*)&Ps[wid][r][16 + hi * 8];

    // O += P * V
#pragma unroll
    for (int ct = 0; ct < 4; ++ct) {
      Frag16 vf;
      vf.h[0] = *(const v8bf*)&Vt[ct * 16 + r][hi * 8];
      vf.h[1] = *(const v8bf*)&Vt[ct * 16 + r][16 + hi * 8];
      o[ct] = WMMA_BF16(pf.v, vf.v, o[ct]);
    }
  }

  // normalize and write into concat-heads layout
#pragma unroll
  for (int ct = 0; ct < 4; ++ct)
#pragma unroll
    for (int i = 0; i < 8; ++i) {
      float val = o[ct][i] / l[i];
      long row = (long)n * SEQ + q0 + hi * 8 + i;
      cat[row * DM + h * DH + ct * 16 + r] = (bf16_t)val;
    }
}

// ============================================================
// residual add + LayerNorm over DM=1024; one block (256 thr) per row.
// Writes f32 (for next residual) and optional bf16 (for next GEMM).
// ============================================================
__global__ __launch_bounds__(256)
void eb_add_ln(const float* __restrict__ a, const float* __restrict__ b,
               const float* __restrict__ g, const float* __restrict__ be,
               float* __restrict__ outF, bf16_t* __restrict__ outB)
{
  const int row = blockIdx.x;
  const long off = (long)row * DM;
  const int tid = threadIdx.x;
  float vreg[4];
  float s = 0.f, ss = 0.f;
#pragma unroll
  for (int i = 0; i < 4; ++i) {
    int c = tid + i * 256;
    float xv = a[off + c] + b[off + c];
    vreg[i] = xv; s += xv; ss += xv * xv;
  }
#pragma unroll
  for (int o2 = 1; o2 < 32; o2 <<= 1) { s += __shfl_xor(s, o2, 32); ss += __shfl_xor(ss, o2, 32); }
  __shared__ float rs[8], rss[8];
  int wid = tid >> 5, lane = tid & 31;
  if (lane == 0) { rs[wid] = s; rss[wid] = ss; }
  __syncthreads();
  if (wid == 0) {
    float s2  = (lane < 8) ? rs[lane]  : 0.f;
    float ss2 = (lane < 8) ? rss[lane] : 0.f;
#pragma unroll
    for (int o2 = 1; o2 < 8; o2 <<= 1) { s2 += __shfl_xor(s2, o2, 32); ss2 += __shfl_xor(ss2, o2, 32); }
    if (lane == 0) { rs[0] = s2; rss[0] = ss2; }
  }
  __syncthreads();
  s = rs[0]; ss = rss[0];
  float mu = s / (float)DM;
  float var = ss / (float)DM - mu * mu;
  float rstd = rsqrtf(var + 1e-5f);
#pragma unroll
  for (int i = 0; i < 4; ++i) {
    int c = tid + i * 256;
    float y = (vreg[i] - mu) * rstd * g[c] + be[c];
    outF[off + c] = y;
    if (outB) outB[off + c] = (bf16_t)y;
  }
}

// ============================================================
// host-side orchestration
// ============================================================
extern "C" void kernel_launch(void* const* d_in, const int* in_sizes, int n_in,
                              void* d_out, int out_size, void* d_ws, size_t ws_size,
                              hipStream_t stream)
{
  const float* x   = (const float*)d_in[0];
  const float* Wq  = (const float*)d_in[1];
  const float* bq  = (const float*)d_in[2];
  const float* Wk  = (const float*)d_in[3];
  const float* bk  = (const float*)d_in[4];
  const float* Wv  = (const float*)d_in[5];
  const float* bv  = (const float*)d_in[6];
  const float* Wo  = (const float*)d_in[7];
  const float* bo  = (const float*)d_in[8];
  const float* W1  = (const float*)d_in[9];
  const float* b1  = (const float*)d_in[10];
  const float* W2  = (const float*)d_in[11];
  const float* b2  = (const float*)d_in[12];
  const float* g1  = (const float*)d_in[13];
  const float* be1 = (const float*)d_in[14];
  const float* g2  = (const float*)d_in[15];
  const float* be2 = (const float*)d_in[16];

  char* ws = (char*)d_ws;
  size_t off = 0;
  auto walloc = [&](size_t bytes) -> void* {
    void* p = ws + off;
    off += (bytes + 255) & ~(size_t)255;
    return p;
  };

  const long NK = (long)NB * SEQ;   // 4096 rows total

  bf16_t* xb   = (bf16_t*)walloc(NK * DM * 2);          // later reused as cat
  bf16_t* qbuf = (bf16_t*)walloc((long)NB * NH * SEQ * DH * 2);
  bf16_t* kbuf = (bf16_t*)walloc((long)NB * NH * SEQ * DH * 2);
  bf16_t* vbuf = (bf16_t*)walloc((long)NB * NH * SEQ * DH * 2);
  bf16_t* Wqb  = (bf16_t*)walloc((long)NH * DM * DH * 2);
  bf16_t* Wkb  = (bf16_t*)walloc((long)NH * DM * DH * 2);
  bf16_t* Wvb  = (bf16_t*)walloc((long)NH * DM * DH * 2);
  bf16_t* Wob  = (bf16_t*)walloc((long)DM * DM * 2);
  bf16_t* W1b  = (bf16_t*)walloc((long)DM * FF * 2);
  bf16_t* W2b  = (bf16_t*)walloc((long)FF * DM * 2);
  float*  attnf= (float*) walloc(NK * DM * 4);          // later reused as ff2
  float*  x1f  = (float*) walloc(NK * DM * 4);
  bf16_t* x1b  = (bf16_t*)walloc(NK * DM * 2);
  bf16_t* ff1b = (bf16_t*)walloc(NK * FF * 2);
  (void)ws_size; (void)in_sizes; (void)n_in; (void)out_size;

  auto cvt = [&](const float* in, bf16_t* out, long n) {
    eb_cvt_bf16<<<dim3(2048), dim3(256), 0, stream>>>(in, out, (int)n);
  };
  cvt(x,  xb,  NK * DM);
  cvt(Wq, Wqb, (long)NH * DM * DH);
  cvt(Wk, Wkb, (long)NH * DM * DH);
  cvt(Wv, Wvb, (long)NH * DM * DH);
  cvt(Wo, Wob, (long)DM * DM);
  cvt(W1, W1b, (long)DM * FF);
  cvt(W2, W2b, (long)FF * DM);

  dim3 blk(256);

  // Q/K/V projections: per (n,h): [2048x1024] x [1024x64] + bias -> bf16
  eb_gemm<64><<<dim3(SEQ/128, 1, NB*NH), blk, 0, stream>>>(
      xb, (long)SEQ * DM, NH, DM,
      Wqb, (long)DM * DH, NH, DH,
      bq, DH,
      nullptr, qbuf, (long)SEQ * DH, DH, DM, 0);
  eb_gemm<64><<<dim3(SEQ/128, 1, NB*NH), blk, 0, stream>>>(
      xb, (long)SEQ * DM, NH, DM,
      Wkb, (long)DM * DH, NH, DH,
      bk, DH,
      nullptr, kbuf, (long)SEQ * DH, DH, DM, 0);
  eb_gemm<64><<<dim3(SEQ/128, 1, NB*NH), blk, 0, stream>>>(
      xb, (long)SEQ * DM, NH, DM,
      Wvb, (long)DM * DH, NH, DH,
      bv, DH,
      nullptr, vbuf, (long)SEQ * DH, DH, DM, 0);

  // attention -> cat (reuses xb storage; xb fully consumed by QKV GEMMs above)
  bf16_t* cat = xb;
  eb_flash<<<dim3(SEQ/128, NB*NH), blk, 0, stream>>>(qbuf, kbuf, vbuf, cat);

  // attn_out = cat @ Wo + bo  (f32)
  eb_gemm<128><<<dim3(NK/128, DM/128, 1), blk, 0, stream>>>(
      cat, 0, 1, DM,
      Wob, 0, 1, DM,
      bo, 0,
      attnf, nullptr, 0, DM, DM, 0);

  // x1 = LN(x + attn_out)
  eb_add_ln<<<dim3((int)NK), blk, 0, stream>>>(x, attnf, g1, be1, x1f, x1b);

  // ff1 = relu(x1 @ W1 + b1) -> bf16
  eb_gemm<128><<<dim3(NK/128, FF/128, 1), blk, 0, stream>>>(
      x1b, 0, 1, DM,
      W1b, 0, 1, FF,
      b1, 0,
      nullptr, ff1b, 0, FF, DM, 1);

  // ff2 = ff1 @ W2 + b2 -> f32 (reuses attnf)
  float* ff2f = attnf;
  eb_gemm<128><<<dim3(NK/128, DM/128, 1), blk, 0, stream>>>(
      ff1b, 0, 1, FF,
      W2b, 0, 1, DM,
      b2, 0,
      ff2f, nullptr, 0, DM, FF, 0);

  // out = LN(x1 + ff2)
  eb_add_ln<<<dim3((int)NK), blk, 0, stream>>>(x1f, ff2f, g2, be2, (float*)d_out, nullptr);
}
